// PairWiseAttention_76871324664262
// MI455X (gfx1250) — matrix operand order repaired and checked by reading, not compiled, not run on hardware
//
#include <hip/hip_runtime.h>

typedef __bf16 bf16_t;
typedef __attribute__((ext_vector_type(16))) __bf16 v16bf;
typedef __attribute__((ext_vector_type(8)))  float  v8f;
typedef __attribute__((ext_vector_type(4)))  float  f32x4;
typedef __attribute__((ext_vector_type(4)))  unsigned int u32x4;
typedef __attribute__((ext_vector_type(2)))  unsigned int u32x2;
typedef __attribute__((ext_vector_type(8)))  int i32x8;
typedef __attribute__((ext_vector_type(4)))  int i32x4;

#define BSIZE 65536
#define DDIM  768
#define NHEAD 12
#define HWIDE 64    // head width = 2*DK
#define BM    128   // rows per workgroup
#define KS    32    // K per bf16 WMMA
#define NKSTEP (DDIM / KS)   // 24

// LDS staging: rows are 64B of data padded to 80B by the TDM pad feature
// (conflict-free ds_load_b128: n*20 mod 64 distinct for n=0..15).
#define ROWB   80
#define ABYTES (BM * ROWB)            // 10240
#define BBYTES (3 * HWIDE * ROWB)     // 15360
#define BUFB   (ABYTES + BBYTES)      // 25600
#define NBUF   3                      // triple-buffered TDM pipeline

#define SGB __builtin_amdgcn_sched_group_barrier

union ABFrag { v16bf v; u32x4 u[2]; };

__device__ __forceinline__ unsigned short f2bf(float f) {
  unsigned u = __float_as_uint(f);
  return (unsigned short)((u + 0x7FFFu + ((u >> 16) & 1u)) >> 16);  // RNE
}

__device__ __forceinline__ float hsum16(float v) {
  v += __shfl_xor(v, 1, 32);
  v += __shfl_xor(v, 2, 32);
  v += __shfl_xor(v, 4, 32);
  v += __shfl_xor(v, 8, 32);
  return v;
}
__device__ __forceinline__ float hsum32(float v) {
  v = hsum16(v);
  v += __shfl_xor(v, 16, 32);
  return v;
}

// ---------------------------------------------------------------------------
// TDM: 2D tile load Global->LDS. 64B-wide tile rows (32 bf16), row stride
// 1536B (= one D-length row), data_size = 8B units, LDS pad 16B per 64B row.
// Descriptor packing per CDNA5 ISA (6-arg form).
// ---------------------------------------------------------------------------
__device__ __forceinline__ void tdm_load_2d(unsigned lds_off, const void* gaddr,
                                            unsigned tensor_rows,
                                            unsigned tile_rows) {
  unsigned long long ga = (unsigned long long)gaddr;
  u32x4 g0;
  g0[0] = 1u;                                   // count=1, user mode
  g0[1] = lds_off;                              // lds_addr (bytes)
  g0[2] = (unsigned)ga;                         // global_addr[31:0]
  g0[3] = (unsigned)(ga >> 32) | (2u << 30);    // global_addr[56:32] | type=2
  i32x8 g1;
  // wg_mask=0 | data_size=3 (8B) | pad_enable | pad_interval=3 (16 dw = 64B)
  // | pad_amount=3 (4 dw = 16B)
  g1[0] = (3 << 16) | (1 << 20) | (3 << 22) | (3 << 25);
  g1[1] = (int)(192u << 16);                    // tensor_dim0[15:0] = 192 units
  g1[2] = (int)((tensor_rows & 0xFFFFu) << 16); // dim0 hi=0 | tensor_dim1 lo
  g1[3] = (int)((tensor_rows >> 16) | (8u << 16)); // tensor_dim1 hi | tile_dim0=8
  g1[4] = (int)(tile_rows & 0xFFFFu);           // tile_dim1 | tile_dim2=0
  g1[5] = 192;                                  // tensor_dim0_stride = 192 units
  g1[6] = 0;
  g1[7] = 0;
  i32x4 z4 = {0, 0, 0, 0};
  i32x8 z8 = {0, 0, 0, 0, 0, 0, 0, 0};
  __builtin_amdgcn_tensor_load_to_lds(g0, g1, z4, z4, z8, 0);
}

// ---------------------------------------------------------------------------
// Kernel 1: fp32 weights -> bf16, 4 elements/thread (b128 load, b64 store).
// ---------------------------------------------------------------------------
__global__ void cvt_w_kernel(const float* __restrict__ wq,
                             const float* __restrict__ wk,
                             const float* __restrict__ wv,
                             unsigned short* __restrict__ out) {
  const float* src = (blockIdx.y == 0) ? wq : (blockIdx.y == 1) ? wk : wv;
  size_t i4 = ((size_t)blockIdx.x * blockDim.x + threadIdx.x) * 4;
  f32x4 v = *(const f32x4*)(src + i4);
  u32x2 o;
  o[0] = (unsigned)f2bf(v[0]) | ((unsigned)f2bf(v[1]) << 16);
  o[1] = (unsigned)f2bf(v[2]) | ((unsigned)f2bf(v[3]) << 16);
  *(u32x2*)(out + (size_t)blockIdx.y * (DDIM * DDIM) + i4) = o;
}

// ---------------------------------------------------------------------------
// Kernel 2: LayerNorm (fp32) -> bf16. One wave per row, b128 loads,
// packed b64 stores. block = 256 (8 waves), grid = BSIZE/8.
// ---------------------------------------------------------------------------
__global__ void ln_kernel(const float* __restrict__ x,
                          const float* __restrict__ gamma,
                          const float* __restrict__ beta,
                          unsigned short* __restrict__ xn) {
  const int lane = threadIdx.x & 31;
  const size_t row = (size_t)blockIdx.x * 8 + (threadIdx.x >> 5);
  const float* xr = x + row * DDIM;

  f32x4 v[6];
  float s = 0.f;
#pragma unroll
  for (int j = 0; j < 6; ++j) {
    v[j] = *(const f32x4*)(xr + j * 128 + lane * 4);
    s += (v[j][0] + v[j][1]) + (v[j][2] + v[j][3]);
  }
  s = hsum32(s);
  const float mu = s * (1.0f / DDIM);
  float s2 = 0.f;
#pragma unroll
  for (int j = 0; j < 6; ++j) {
#pragma unroll
    for (int q = 0; q < 4; ++q) { float d = v[j][q] - mu; s2 += d * d; }
  }
  s2 = hsum32(s2);
  const float inv = rsqrtf(s2 * (1.0f / DDIM) + 1e-6f);

  unsigned short* o = xn + row * DDIM;
#pragma unroll
  for (int j = 0; j < 6; ++j) {
    const int c = j * 128 + lane * 4;
    const f32x4 g = *(const f32x4*)(gamma + c);
    const f32x4 b = *(const f32x4*)(beta + c);
    const unsigned short y0 = f2bf((v[j][0] - mu) * inv * g[0] + b[0]);
    const unsigned short y1 = f2bf((v[j][1] - mu) * inv * g[1] + b[1]);
    const unsigned short y2 = f2bf((v[j][2] - mu) * inv * g[2] + b[2]);
    const unsigned short y3 = f2bf((v[j][3] - mu) * inv * g[3] + b[3]);
    u32x2 ov;
    ov[0] = (unsigned)y0 | ((unsigned)y1 << 16);
    ov[1] = (unsigned)y2 | ((unsigned)y3 << 16);
    *(u32x2*)(o + c) = ov;
  }
}

// ---------------------------------------------------------------------------
// Kernel 3: fused bf16 GEMM (Q,K,V) + per-head 2x2 attention + residual.
// grid: (BSIZE/BM, NHEAD), block 256 = 8 waves.
// Triple-buffered TDM staging (issue for s+2 while computing s); inner loop
// is ds_load_b128 + v_wmma with a sched_group_barrier-enforced 1-fragment
// software pipeline (waits become s_wait_dscnt 2 instead of 0).
// ---------------------------------------------------------------------------
__global__ void __launch_bounds__(256)
gemm_attn_kernel(const bf16_t* __restrict__ xn,
                 const bf16_t* __restrict__ wq,
                 const bf16_t* __restrict__ wk,
                 const bf16_t* __restrict__ wv,
                 const float* __restrict__ x,
                 float* __restrict__ out) {
  __shared__ __attribute__((aligned(128))) unsigned char smem[NBUF][BUFB];

  const int lane = threadIdx.x & 31;
  const int wave = threadIdx.x >> 5;
  const int head = blockIdx.y;
  const int row0 = blockIdx.x * BM + wave * 16;
  const int r = lane & 15;   // row (A) / column (B) within tile
  const int h = lane >> 4;   // lane half selects K sub-range

  v8f acc[3][4];
  const v8f vzero = {0.f, 0.f, 0.f, 0.f, 0.f, 0.f, 0.f, 0.f};
#pragma unroll
  for (int m = 0; m < 3; ++m)
#pragma unroll
    for (int t = 0; t < 4; ++t) acc[m][t] = vzero;

  const bf16_t* W[3] = {wq, wk, wv};
  const size_t aRowOff = (size_t)(blockIdx.x * BM) * DDIM;  // elements
  const size_t bRowOff = (size_t)(head * HWIDE) * DDIM;     // elements

  // LDS byte offsets of the staging buffers (low 32 bits of the generic
  // shared-memory address are the LDS offset).
  const unsigned ldsBase = (unsigned)(unsigned long long)(void*)&smem[0][0];
  const unsigned ldsBuf[NBUF] = {ldsBase, ldsBase + BUFB, ldsBase + 2 * BUFB};

  // Prologue: stage k-steps 0 and 1 (two 4-op in-order TDM batches).
  if (wave == 0) {
#pragma unroll
    for (int p = 0; p < 2; ++p) {
      const int kp = p * KS;
      tdm_load_2d(ldsBuf[p], xn + aRowOff + kp, BSIZE, BM);
#pragma unroll
      for (int m = 0; m < 3; ++m)
        tdm_load_2d(ldsBuf[p] + ABYTES + m * (HWIDE * ROWB),
                    W[m] + bRowOff + kp, DDIM, HWIDE);
    }
  }

#pragma unroll 1
  for (int s = 0; s < NKSTEP; ++s) {
    const int cur = s % NBUF;

    __syncthreads();  // consumers of buf[(s+2)%3] (used at s-1) are done
    if (wave == 0) {
      if (s + 2 < NKSTEP) {
        const int pre = (s + 2) % NBUF;
        const int k2 = (s + 2) * KS;
        tdm_load_2d(ldsBuf[pre], xn + aRowOff + k2, BSIZE, BM);
#pragma unroll
        for (int m = 0; m < 3; ++m)
          tdm_load_2d(ldsBuf[pre] + ABYTES + m * (HWIDE * ROWB),
                      W[m] + bRowOff + k2, DDIM, HWIDE);
        __builtin_amdgcn_s_wait_tensorcnt(8);  // batch s done (in-order TDM)
      } else if (s + 1 < NKSTEP) {
        __builtin_amdgcn_s_wait_tensorcnt(4);
      } else {
        __builtin_amdgcn_s_wait_tensorcnt(0);
      }
    }
    __syncthreads();  // buf[cur] ready for everyone

    // A fragment 16x32 bf16 from LDS: lane r -> K{0..15} (bytes
    // h*32..h*32+31 of its 64B row), lane r+16 -> K{16..31}.
    const unsigned char* sbase = &smem[cur][0];
    const unsigned char* abase = sbase + (wave * 16 + r) * ROWB + h * 32;
    ABFrag a;
    a.u[0] = *(const u32x4*)(abase);
    a.u[1] = *(const u32x4*)(abase + 16);

    // B fragments 32x16: lane n holds column (t*16+n), K-half h -> 32
    // contiguous bytes at row offset h*32. 1-deep register pipeline; issue
    // order pinned below with sched_group_barrier.
    const unsigned char* bcol = sbase + ABYTES + r * ROWB + h * 32;
    ABFrag bf[2];
    bf[0].u[0] = *(const u32x4*)(bcol);
    bf[0].u[1] = *(const u32x4*)(bcol + 16);
#pragma unroll
    for (int i = 0; i < 12; ++i) {
      if (i + 1 < 12) {
        const int m1 = (i + 1) >> 2, t1 = (i + 1) & 3;
        const unsigned char* nb =
            bcol + m1 * (HWIDE * ROWB) + t1 * (16 * ROWB);
        bf[(i + 1) & 1].u[0] = *(const u32x4*)(nb);
        bf[(i + 1) & 1].u[1] = *(const u32x4*)(nb + 16);
      }
      const int m = i >> 2, t = i & 3;
      acc[m][t] = __builtin_amdgcn_wmma_f32_16x16x32_bf16(
          false, a.v, false, bf[i & 1].v, (short)0, acc[m][t], false, false);
    }

    // Enforce software pipeline: 6 DS reads (A + B0 + B1), then
    // (1 WMMA, 2 DS reads) x10, then the last 2 WMMAs.
    SGB(0x100, 6, 0);                       // DS reads
    SGB(0x008, 1, 0); SGB(0x100, 2, 0);
    SGB(0x008, 1, 0); SGB(0x100, 2, 0);
    SGB(0x008, 1, 0); SGB(0x100, 2, 0);
    SGB(0x008, 1, 0); SGB(0x100, 2, 0);
    SGB(0x008, 1, 0); SGB(0x100, 2, 0);
    SGB(0x008, 1, 0); SGB(0x100, 2, 0);
    SGB(0x008, 1, 0); SGB(0x100, 2, 0);
    SGB(0x008, 1, 0); SGB(0x100, 2, 0);
    SGB(0x008, 1, 0); SGB(0x100, 2, 0);
    SGB(0x008, 1, 0); SGB(0x100, 2, 0);
    SGB(0x008, 2, 0);                       // final 2 WMMAs
  }

  // Epilogue: per-head 2x2 attention entirely in registers.
  // C layout: VGPR e, lanes 0-15 -> row e, lanes 16-31 -> row e+8; N = t*16+r.
  const float sscale = 0.17677669529663689f;  // 1/sqrt(32)
#pragma unroll
  for (int e = 0; e < 8; ++e) {
    float s00 = hsum16(acc[0][0][e] * acc[1][0][e] + acc[0][1][e] * acc[1][1][e]);
    float s01 = hsum16(acc[0][0][e] * acc[1][2][e] + acc[0][1][e] * acc[1][3][e]);
    float s10 = hsum16(acc[0][2][e] * acc[1][0][e] + acc[0][3][e] * acc[1][1][e]);
    float s11 = hsum16(acc[0][2][e] * acc[1][2][e] + acc[0][3][e] * acc[1][3][e]);
    s00 *= sscale; s01 *= sscale; s10 *= sscale; s11 *= sscale;
    const float m0 = fmaxf(s00, s01), m1 = fmaxf(s10, s11);
    const float e00 = __expf(s00 - m0), e01 = __expf(s01 - m0);
    const float e10 = __expf(s10 - m1), e11 = __expf(s11 - m1);
    const float d0 = 1.f / (e00 + e01), d1 = 1.f / (e10 + e11);
    const float a00 = e00 * d0, a01 = e01 * d0;
    const float a10 = e10 * d1, a11 = e11 * d1;

    const int rowg = row0 + e + 8 * h;
    const size_t base = (size_t)rowg * DDIM + head * HWIDE + r;
    const float o0 = a00 * acc[2][0][e] + a01 * acc[2][2][e];  // p=0, d 0..15
    const float o1 = a00 * acc[2][1][e] + a01 * acc[2][3][e];  // p=0, d 16..31
    const float o2 = a10 * acc[2][0][e] + a11 * acc[2][2][e];  // p=1, d 0..15
    const float o3 = a10 * acc[2][1][e] + a11 * acc[2][3][e];  // p=1, d 16..31
    out[base + 0]  = o0 + x[base + 0];
    out[base + 16] = o1 + x[base + 16];
    out[base + 32] = o2 + x[base + 32];
    out[base + 48] = o3 + x[base + 48];
  }
}

// ---------------------------------------------------------------------------
extern "C" void kernel_launch(void* const* d_in, const int* in_sizes, int n_in,
                              void* d_out, int out_size, void* d_ws, size_t ws_size,
                              hipStream_t stream) {
  const float* x     = (const float*)d_in[0];
  const float* Wq    = (const float*)d_in[1];
  const float* Wk    = (const float*)d_in[2];
  const float* Wv    = (const float*)d_in[3];
  const float* gamma = (const float*)d_in[4];
  const float* beta  = (const float*)d_in[5];
  float* out = (float*)d_out;

  // Workspace layout (bf16 as u16): Wq | Wk | Wv | xn   (~104 MB total)
  unsigned short* wq_b = (unsigned short*)d_ws;
  unsigned short* wk_b = wq_b + (size_t)DDIM * DDIM;
  unsigned short* wv_b = wk_b + (size_t)DDIM * DDIM;
  unsigned short* xn   = wv_b + (size_t)DDIM * DDIM;

  cvt_w_kernel<<<dim3(DDIM * DDIM / 4 / 256, 3), 256, 0, stream>>>(Wq, Wk, Wv,
                                                                   wq_b);
  ln_kernel<<<BSIZE / 8, 256, 0, stream>>>(x, gamma, beta, xn);
  gemm_attn_kernel<<<dim3(BSIZE / BM, NHEAD), 256, 0, stream>>>(
      (const bf16_t*)xn, (const bf16_t*)wq_b, (const bf16_t*)wk_b,
      (const bf16_t*)wv_b, x, out);
}
